// MeanPooling_50869592654341
// MI455X (gfx1250) — compile-verified, matrix operand order checked by reading, not run on hardware
//
#include <hip/hip_runtime.h>

// -------- problem constants (from reference) --------
#define B_  8
#define E_  512
#define L_  4096
#define D_  256

// -------- tiling --------
#define TM   128      // C-tile rows per workgroup (E dim)
#define TN   64       // C-tile cols per workgroup (D dim)
#define TK   32       // K step (matches v_wmma_f32_16x16x32_bf16)
#define LDA  40       // LDS row stride in halfs (32 + 8 pad) -> 80 bytes, 16B aligned
#define LDB  40

typedef __attribute__((ext_vector_type(16))) __bf16 v16bf;
typedef __attribute__((ext_vector_type(8)))  __bf16 v8bf;
typedef __attribute__((ext_vector_type(8)))  float  v8f;
typedef __attribute__((ext_vector_type(4)))  float  v4f;

__global__ __launch_bounds__(256) void mean_pool_wmma_bf16(
    const float* __restrict__ doc,   // [B, L, D]
    const float* __restrict__ emap,  // [B, E, L]
    const float* __restrict__ lens,  // [B, E]
    float* __restrict__ out)         // [B, E, D]
{
    __shared__ __align__(16) __bf16 Alds[2][TM][LDA];      // [row][k]   bf16
    __shared__ __align__(16) __bf16 Blds[2][TN][LDB];      // [col][k]   bf16 (transposed)

    const int tid  = threadIdx.x;
    const int lane = tid & 31;
    const int wave = tid >> 5;
    const int wrow = wave >> 1;          // 0..3 -> 32-row strip
    const int wcol = wave & 1;           // 0..1 -> 32-col strip

    // fragment lane decomposition (documented 16-bit WMMA layouts)
    const int kb = (lane >> 4) << 3;     // 0 or 8 (K sub-block select)
    const int ml = lane & 15;            // row (A) / col (B) within 16

    const int t  = blockIdx.x;           // 128 tiles total
    const int b  = t >> 4;
    const int m0 = ((t >> 2) & 3) * TM;
    const int n0 = (t & 3) * TN;

    const float* Ag = emap + (size_t)b * E_ * L_ + (size_t)m0 * L_;  // row-major [128][L]
    const float* Bg = doc  + (size_t)b * L_ * D_ + n0;               // row-major [L][64-slice]

    // staging registers for the next K tile (global -> regs -> bf16 -> LDS)
    v4f ast[4];   // A tile: 128x32 f32 = 1024 float4, 4 per thread
    v4f bst[2];   // B tile:  32x64 f32; each thread: 2 adjacent k-rows x 4 cols

    // B cooperative-load coordinates: thread covers k-rows {bkr, bkr+1}, cols [bcq, bcq+3]
    const int bkr = (tid >> 4) << 1;     // 0,2,...,30
    const int bcq = (tid & 15) << 2;     // 0,4,...,60

    auto load_tile = [&](int k0) {
#pragma unroll
        for (int i = 0; i < 4; ++i) {
            const int f   = tid + i * 256;
            const int row = f >> 3;                 // 8 float4 per 32-wide row
            const int kq  = (f & 7) << 2;
            ast[i] = *(const v4f*)(Ag + (size_t)row * L_ + (k0 + kq));
        }
        bst[0] = *(const v4f*)(Bg + (size_t)(k0 + bkr)     * D_ + bcq);
        bst[1] = *(const v4f*)(Bg + (size_t)(k0 + bkr + 1) * D_ + bcq);
    };

    auto store_tile = [&](int buf) {
#pragma unroll
        for (int i = 0; i < 4; ++i) {
            const int f   = tid + i * 256;
            const int row = f >> 3;
            const int kq  = (f & 7) << 2;
            __bf16* p = &Alds[buf][row][kq];        // consecutive k -> ds_store_b64
            p[0] = (__bf16)ast[i].x;  p[1] = (__bf16)ast[i].y;
            p[2] = (__bf16)ast[i].z;  p[3] = (__bf16)ast[i].w;
        }
        // transpose into [col][k]; (k, k+1) are consecutive halfs -> cvt_pk + ds_store_b32
#pragma unroll
        for (int c = 0; c < 4; ++c) {
            __bf16* q = &Blds[buf][bcq + c][bkr];
            q[0] = (__bf16)bst[0][c];
            q[1] = (__bf16)bst[1][c];
        }
    };

    const v8f vzero = {0.f, 0.f, 0.f, 0.f, 0.f, 0.f, 0.f, 0.f};
    v8f acc[2][2];
#pragma unroll
    for (int i = 0; i < 2; ++i)
#pragma unroll
        for (int j = 0; j < 2; ++j) acc[i][j] = vzero;

    // prologue: tile0 -> LDS buf0; tile1 staged in regs
    load_tile(0);
    store_tile(0);
    load_tile(TK);
    __syncthreads();

    const int NK = L_ / TK;   // 128 K-steps
    for (int kt = 0; kt < NK; ++kt) {
        const int cur = kt & 1;

        if (kt + 3 < NK) {
            // warm GL2 well ahead (global_prefetch_b8)
            const int k3 = (kt + 3) * TK;
            __builtin_prefetch(Ag + (size_t)(tid >> 1) * L_ + k3 + (tid & 1) * 16, 0, 1);
            __builtin_prefetch(Bg + (size_t)(k3 + (tid & 31)) * D_ + (tid >> 5) * 8, 0, 1);
        }

        // ---- build fragments from LDS per documented 16-bit WMMA layouts ----
        // A (16x32, MxK): lanes 0-15 hold M=lane, K 0-7 (v0-3) and 16-23 (v4-7);
        //                 lanes 16-31 hold M=lane-16, K 8-15 and 24-31.
        v16bf afrag[2], bfrag[2];
#pragma unroll
        for (int i = 0; i < 2; ++i) {
            const __bf16* ap = &Alds[cur][wrow * 32 + i * 16 + ml][kb];
            v8bf lo = *(const v8bf*)(ap);        // K = kb .. kb+7
            v8bf hi = *(const v8bf*)(ap + 16);   // K = kb+16 .. kb+23
            afrag[i] = __builtin_shufflevector(lo, hi,
                0, 1, 2, 3, 4, 5, 6, 7, 8, 9, 10, 11, 12, 13, 14, 15);
        }
#pragma unroll
        for (int j = 0; j < 2; ++j) {
            const __bf16* bp = &Blds[cur][wcol * 32 + j * 16 + ml][kb];
            v8bf lo = *(const v8bf*)(bp);
            v8bf hi = *(const v8bf*)(bp + 16);
            bfrag[j] = __builtin_shufflevector(lo, hi,
                0, 1, 2, 3, 4, 5, 6, 7, 8, 9, 10, 11, 12, 13, 14, 15);
        }

#pragma unroll
        for (int i = 0; i < 2; ++i)
#pragma unroll
            for (int j = 0; j < 2; ++j)
                acc[i][j] = __builtin_amdgcn_wmma_f32_16x16x32_bf16(
                    /*neg_a=*/false, afrag[i],
                    /*neg_b=*/false, bfrag[j],
                    /*c_mod=*/(short)0, acc[i][j],
                    /*reuse_a=*/false, /*reuse_b=*/false);

        // stage tile kt+1 (already in regs) into the other buffer; no conflict
        // with concurrent readers of `cur`, so a single barrier suffices.
        if (kt + 1 < NK) store_tile(cur ^ 1);
        // start global loads for tile kt+2 (consumed two iterations later)
        if (kt + 2 < NK) load_tile((kt + 2) * TK);

        __syncthreads();
    }

    // ---- epilogue: scale rows by 1/len and store ----
    // C layout (16x16 f32): VGPR r -> M = r (+8 for lanes 16-31), N = lane&15.
    const float* lensb = lens + (size_t)b * E_ + m0;
    float* Og = out + (size_t)b * E_ * D_ + (size_t)m0 * D_ + n0;
    const int mhi = (lane >> 4) << 3;   // 0 or 8

#pragma unroll
    for (int i = 0; i < 2; ++i) {
#pragma unroll
        for (int j = 0; j < 2; ++j) {
            const int mbase = wrow * 32 + i * 16 + mhi;
            const int ncol  = wcol * 32 + j * 16 + ml;
#pragma unroll
            for (int r = 0; r < 8; ++r) {
                const int m = mbase + r;
                const float inv = 1.0f / lensb[m];
                Og[(size_t)m * D_ + ncol] = acc[i][j][r] * inv;
            }
        }
    }
}

extern "C" void kernel_launch(void* const* d_in, const int* in_sizes, int n_in,
                              void* d_out, int out_size, void* d_ws, size_t ws_size,
                              hipStream_t stream) {
    (void)in_sizes; (void)n_in; (void)d_ws; (void)ws_size; (void)out_size;
    const float* doc  = (const float*)d_in[0];   // doc_state      [B,L,D]
    const float* emap = (const float*)d_in[1];   // entity_mapping [B,E,L]
    const float* lens = (const float*)d_in[2];   // entity_lens    [B,E]
    float* out = (float*)d_out;                  // [B,E,D] fp32

    const int grid = B_ * (E_ / TM) * (D_ / TN); // 8*4*4 = 128 workgroups
    mean_pool_wmma_bf16<<<grid, 256, 0, stream>>>(doc, emap, lens, out);
}
